// MPNNModel_8942121910898
// MI455X (gfx1250) — matrix operand order repaired and checked by reading, not compiled, not run on hardware
//
#include <hip/hip_runtime.h>

#define NN    50000
#define NE    1600000
#define LAYERS 4
#define BN_EPS 1e-5f

typedef __attribute__((ext_vector_type(16))) __bf16 v16bf;
typedef __attribute__((ext_vector_type(8)))  float  v8f;

union FragU {
  v16bf h;
  unsigned int w[8];
};

__device__ __forceinline__ unsigned short f2bf(float f) {
  union { float f; unsigned u; } v; v.f = f;
  unsigned r = v.u + 0x7FFFu + ((v.u >> 16) & 1u);   // round-to-nearest-even
  return (unsigned short)(r >> 16);
}
__device__ __forceinline__ float bf2f(unsigned short s) {
  union { unsigned u; float f; } v; v.u = ((unsigned)s) << 16;
  return v.f;
}
__device__ __forceinline__ unsigned int pack2(float a, float b) {
  return (unsigned)f2bf(a) | ((unsigned)f2bf(b) << 16);
}

__device__ __forceinline__ v8f wmma_bf16(v16bf a, v16bf b, v8f c) {
  // D = A(16x32 bf16) * B(32x16 bf16) + C(16x16 f32)
  return __builtin_amdgcn_wmma_f32_16x16x32_bf16(false, a, false, b, (short)0, c,
                                                 false, false);
}

// Async global -> LDS copy, 16B per lane, tracked by ASYNCcnt (ISA 08 §4, GV mode)
__device__ __forceinline__ void async_copy_b128(unsigned lds_byte_off,
                                                const void* gaddr) {
  asm volatile("global_load_async_to_lds_b128 %0, %1, off"
               :: "v"(lds_byte_off), "v"(gaddr)
               : "memory");
}
__device__ __forceinline__ void wait_async0() {
  asm volatile("s_wait_asynccnt 0x0" ::: "memory");
}
__device__ __forceinline__ unsigned lds_off32(const void* p) {
  return (unsigned)(unsigned long long)p;   // low 32 bits of flat shared addr
}

// A fragment (16x32 bf16) from LDS tile. ISA layout: lanes 0-15 -> M=lane,
// lanes 16-31 same M with K offset +8; VGPR p<4 covers K 2p..2p+1 (+8*half),
// p>=4 covers K 16+2(p-4)..+1 (+8*half).
__device__ __forceinline__ v16bf load_A_lds(const unsigned short* lds, int stride,
                                            int kbase, int lane) {
  int m = lane & 15, half = lane >> 4;
  FragU f;
#pragma unroll
  for (int p = 0; p < 8; ++p) {
    int kp = (p < 4) ? (2 * p + 8 * half) : (2 * p + 8 + 8 * half);
    f.w[p] = *(const unsigned int*)(lds + m * stride + kbase + kp);
  }
  return f.h;
}

// Same, applying BatchNorm affine (per-K scale/shift) + ReLU during conversion.
__device__ __forceinline__ v16bf load_A_bnrelu(const unsigned short* lds, int stride,
                                               int kbase, int lane,
                                               const float* sc, const float* sh) {
  int m = lane & 15, half = lane >> 4;
  FragU f;
#pragma unroll
  for (int p = 0; p < 8; ++p) {
    int kp = (p < 4) ? (2 * p + 8 * half) : (2 * p + 8 + 8 * half);
    int k = kbase + kp;
    unsigned int raw = *(const unsigned int*)(lds + m * stride + k);
    float x0 = bf2f((unsigned short)(raw & 0xffffu));
    float x1 = bf2f((unsigned short)(raw >> 16));
    float y0 = fmaxf(0.f, x0 * sc[k] + sh[k]);
    float y1 = fmaxf(0.f, x1 * sc[k + 1] + sh[k + 1]);
    f.w[p] = pack2(y0, y1);
  }
  return f.h;
}

// B fragment (32x16 bf16): lane%16 = N column, K in same order as A fragment.
__device__ __forceinline__ v16bf load_B(const unsigned short* W, int kglob_base,
                                        int lane, int n0) {
  int n = n0 + (lane & 15), half = lane >> 4;
  FragU f;
#pragma unroll
  for (int p = 0; p < 8; ++p) {
    int kp = (p < 4) ? (2 * p + 8 * half) : (2 * p + 8 + 8 * half);
    const unsigned short* c = W + (long)(kglob_base + kp) * 64 + n;
    f.w[p] = (unsigned)c[0] | ((unsigned)c[64] << 16);
  }
  return f.h;
}

// ---------------------------------------------------------------------------
__global__ __launch_bounds__(256) void cvt_f32_bf16(const float* __restrict__ in,
                                                    unsigned short* __restrict__ out,
                                                    int n) {
  int i = blockIdx.x * 256 + threadIdx.x;
  if (i < n) out[i] = f2bf(in[i]);
}

// h = x @ lin_in_W + b ; f32 master + bf16 copy for WMMA gathers
__global__ __launch_bounds__(256) void lin_in_kernel(const float* __restrict__ x,
                                                     const float* __restrict__ W,
                                                     const float* __restrict__ b,
                                                     float* __restrict__ h,
                                                     unsigned short* __restrict__ hbf) {
  long gid = (long)blockIdx.x * 256 + threadIdx.x;
  if (gid >= (long)NN * 64) return;
  int v = (int)(gid >> 6), c = (int)(gid & 63);
  float acc = b[c];
#pragma unroll
  for (int k = 0; k < 6; ++k) acc += x[v * 6 + k] * W[k * 64 + c];
  h[gid] = acc;
  hbf[gid] = f2bf(acc);
}

// Edge GEMM1: z1 = [h[dst] | h[src] | ea] @ W1 + b1.  64-edge tiles, 4 waves,
// each wave owns a 16-col N tile and runs 4 M-subtiles x 4 K-slices = 16 WMMA.
__global__ __launch_bounds__(128) void edge_msg_gemm1(
    const unsigned short* __restrict__ hbf, const int* __restrict__ edge_index,
    const float* __restrict__ edge_attr, const unsigned short* __restrict__ W1bf,
    const float* __restrict__ W1f, const float* __restrict__ b1,
    unsigned short* __restrict__ z1, float* __restrict__ stats) {
  __shared__ __align__(16) unsigned short Atile[64 * 136];
  __shared__ float eaS[64];
  int t = threadIdx.x;
  long e0 = (long)blockIdx.x * 64;
  __builtin_prefetch(W1bf, 0, 1);

  // Stage gathered rows with async global->LDS: one row-half (64 bf16) per thread
  {
    int r = t & 63;
    bool isDst = (t < 64);
    int node = edge_index[(isDst ? (long)NE : 0L) + e0 + r];
    const unsigned short* gsrc = hbf + (long)node * 64;
    unsigned ldsBase = lds_off32(Atile + r * 136 + (isDst ? 0 : 64));
#pragma unroll
    for (int j = 0; j < 8; ++j)
      async_copy_b128(ldsBase + j * 16, gsrc + j * 8);
    if (t < 64) eaS[t] = edge_attr[e0 + t];
  }
  wait_async0();
  __syncthreads();

  int lane = t & 31;
  int n0 = (t >> 5) * 16;
  v16bf Bf[4];
#pragma unroll
  for (int s = 0; s < 4; ++s) Bf[s] = load_B(W1bf, s * 32, lane, n0);

  int n = n0 + (lane & 15);
  int half = lane >> 4;
  float bias = b1[n];
  float w128 = W1f[128 * 64 + n];
  float sum = 0.f, ss = 0.f;
#pragma unroll
  for (int mt = 0; mt < 4; ++mt) {
    v8f acc = {};
#pragma unroll
    for (int s = 0; s < 4; ++s) {
      v16bf a = load_A_lds(Atile + mt * 16 * 136, 136, s * 32, lane);
      acc = wmma_bf16(a, Bf[s], acc);
    }
#pragma unroll
    for (int rr = 0; rr < 8; ++rr) {
      int row = mt * 16 + half * 8 + rr;
      float val = acc[rr] + bias + eaS[row] * w128;
      sum += val; ss += val * val;
      z1[(e0 + row) * 64 + n] = f2bf(val);
    }
  }
  sum += __shfl_xor(sum, 16, 32);   // combine half-lanes sharing column n
  ss  += __shfl_xor(ss, 16, 32);
  if (half == 0) {
    atomicAdd(&stats[n], sum);
    atomicAdd(&stats[64 + n], ss);
  }
}

// Second MLP GEMM with fused BN+ReLU on input (edge and node variants; tail-guarded)
__global__ __launch_bounds__(128) void mlp2_bn_gemm(
    const unsigned short* __restrict__ zin, const float* __restrict__ statsIn,
    const float* __restrict__ gamma, const float* __restrict__ beta, float invCnt,
    const unsigned short* __restrict__ Wbf, const float* __restrict__ bias,
    unsigned short* __restrict__ zout, float* __restrict__ statsOut, long M) {
  __shared__ __align__(16) unsigned short Zt[64 * 72];
  __shared__ float sc[64], sh[64];
  int t = threadIdx.x;
  long r0 = (long)blockIdx.x * 64;
  long rem = M - r0;
  int rows = rem > 64 ? 64 : (int)rem;
  __builtin_prefetch(Wbf, 0, 1);

  {
    int r = t >> 1, c = (t & 1) * 32;     // each thread copies half a row (64B)
    if (r < rows) {
      const unsigned short* gsrc = zin + (r0 + r) * 64 + c;
      unsigned ldsBase = lds_off32(Zt + r * 72 + c);
#pragma unroll
      for (int j = 0; j < 4; ++j)
        async_copy_b128(ldsBase + j * 16, gsrc + j * 8);
    }
  }
  if (t < 64) {
    float mu  = statsIn[t] * invCnt;
    float var = statsIn[64 + t] * invCnt - mu * mu;
    float rs  = rsqrtf(var + BN_EPS);
    float g   = gamma[t];
    sc[t] = rs * g;
    sh[t] = beta[t] - mu * rs * g;
  }
  wait_async0();
  __syncthreads();

  int lane = t & 31;
  int n0 = (t >> 5) * 16;
  v16bf Bf[2];
#pragma unroll
  for (int s = 0; s < 2; ++s) Bf[s] = load_B(Wbf, s * 32, lane, n0);

  int n = n0 + (lane & 15);
  int half = lane >> 4;
  float bv = bias[n];
  float sum = 0.f, ss = 0.f;
#pragma unroll
  for (int mt = 0; mt < 4; ++mt) {
    v8f acc = {};
#pragma unroll
    for (int s = 0; s < 2; ++s) {
      v16bf a = load_A_bnrelu(Zt + mt * 16 * 72, 72, s * 32, lane, sc, sh);
      acc = wmma_bf16(a, Bf[s], acc);
    }
#pragma unroll
    for (int rr = 0; rr < 8; ++rr) {
      int row = mt * 16 + half * 8 + rr;
      if (row < rows) {
        float val = acc[rr] + bv;
        sum += val; ss += val * val;
        zout[(r0 + row) * 64 + n] = f2bf(val);
      }
    }
  }
  sum += __shfl_xor(sum, 16, 32);
  ss  += __shfl_xor(ss, 16, 32);
  if (half == 0) {
    atomicAdd(&statsOut[n], sum);
    atomicAdd(&statsOut[64 + n], ss);
  }
}

// BN+ReLU on z2, segment-sum into aggr via f32 atomics (aggr stays L2-resident)
__global__ __launch_bounds__(256) void bn_relu_scatter(
    const unsigned short* __restrict__ z2, const float* __restrict__ stats,
    const float* __restrict__ gamma, const float* __restrict__ beta, float invCnt,
    const int* __restrict__ edge_index, float* __restrict__ aggr) {
  __shared__ float sc[64], sh[64];
  int t = threadIdx.x;
  if (t < 64) {
    float mu  = stats[t] * invCnt;
    float var = stats[64 + t] * invCnt - mu * mu;
    float rs  = rsqrtf(var + BN_EPS);
    float g   = gamma[t];
    sc[t] = rs * g;
    sh[t] = beta[t] - mu * rs * g;
  }
  __syncthreads();
  long gid = (long)blockIdx.x * 256 + t;
  long e = gid >> 2;
  if (e >= NE) return;
  int c0 = (int)(gid & 3) * 16;
  int dn = edge_index[(long)NE + e];
  const uint4* zp = (const uint4*)(z2 + e * 64 + c0);
  uint4 za = zp[0], zb = zp[1];
  unsigned int zr[8] = {za.x, za.y, za.z, za.w, zb.x, zb.y, zb.z, zb.w};
  float* arow = aggr + (long)dn * 64 + c0;
#pragma unroll
  for (int p = 0; p < 8; ++p) {
#pragma unroll
    for (int q = 0; q < 2; ++q) {
      int c = c0 + 2 * p + q;
      float x   = bf2f((unsigned short)((zr[p] >> (16 * q)) & 0xffffu));
      float val = fmaxf(0.f, x * sc[c] + sh[c]);
      atomicAdd(arow + 2 * p + q, val);
    }
  }
}

// Node GEMM1: u1 = [h | aggr] @ upd_W1 + b.  64-node tiles (tail-guarded).
__global__ __launch_bounds__(128) void node_upd_gemm1(
    const unsigned short* __restrict__ hbf, const float* __restrict__ aggr,
    const unsigned short* __restrict__ Wbf, const float* __restrict__ bias,
    unsigned short* __restrict__ u1, float* __restrict__ stats) {
  __shared__ __align__(16) unsigned short Atile[64 * 136];
  int t = threadIdx.x;
  long v0 = (long)blockIdx.x * 64;
  long rem = (long)NN - v0;
  int rows = rem > 64 ? 64 : (int)rem;
  __builtin_prefetch(Wbf, 0, 1);

  if (t < 64) {                       // h half: async copy, one row per thread
    int r = t;
    if (r < rows) {
      const unsigned short* gsrc = hbf + (v0 + r) * 64;
      unsigned ldsBase = lds_off32(Atile + r * 136);
#pragma unroll
      for (int j = 0; j < 8; ++j)
        async_copy_b128(ldsBase + j * 16, gsrc + j * 8);
    }
  } else {                            // aggr half: f32 -> bf16 via registers
    int r = t - 64;
    if (r < rows) {
      const float4* sp = (const float4*)(aggr + (v0 + r) * 64);
#pragma unroll
      for (int j = 0; j < 8; ++j) {
        float4 f0 = sp[2 * j], f1 = sp[2 * j + 1];
        uint4 w;
        w.x = pack2(f0.x, f0.y); w.y = pack2(f0.z, f0.w);
        w.z = pack2(f1.x, f1.y); w.w = pack2(f1.z, f1.w);
        *(uint4*)(Atile + r * 136 + 64 + j * 8) = w;
      }
    }
  }
  wait_async0();
  __syncthreads();

  int lane = t & 31;
  int n0 = (t >> 5) * 16;
  v16bf Bf[4];
#pragma unroll
  for (int s = 0; s < 4; ++s) Bf[s] = load_B(Wbf, s * 32, lane, n0);

  int n = n0 + (lane & 15);
  int half = lane >> 4;
  float bv = bias[n];
  float sum = 0.f, ss = 0.f;
#pragma unroll
  for (int mt = 0; mt < 4; ++mt) {
    v8f acc = {};
#pragma unroll
    for (int s = 0; s < 4; ++s) {
      v16bf a = load_A_lds(Atile + mt * 16 * 136, 136, s * 32, lane);
      acc = wmma_bf16(a, Bf[s], acc);
    }
#pragma unroll
    for (int rr = 0; rr < 8; ++rr) {
      int row = mt * 16 + half * 8 + rr;
      if (row < rows) {
        float val = acc[rr] + bv;
        sum += val; ss += val * val;
        u1[(v0 + row) * 64 + n] = f2bf(val);
      }
    }
  }
  sum += __shfl_xor(sum, 16, 32);
  ss  += __shfl_xor(ss, 16, 32);
  if (half == 0) {
    atomicAdd(&stats[n], sum);
    atomicAdd(&stats[64 + n], ss);
  }
}

// BN+ReLU on u2, residual h += u, refresh bf16 copy
__global__ __launch_bounds__(256) void bn_relu_residual(
    const unsigned short* __restrict__ u2, const float* __restrict__ stats,
    const float* __restrict__ gamma, const float* __restrict__ beta, float invCnt,
    float* __restrict__ h, unsigned short* __restrict__ hbf) {
  long gid = (long)blockIdx.x * 256 + threadIdx.x;
  if (gid >= (long)NN * 64) return;
  int c = (int)(gid & 63);
  float mu  = stats[c] * invCnt;
  float var = stats[64 + c] * invCnt - mu * mu;
  float rs  = rsqrtf(var + BN_EPS);
  float x   = bf2f(u2[gid]);
  float u   = fmaxf(0.f, (x - mu) * rs * gamma[c] + beta[c]);
  float nh  = h[gid] + u;
  h[gid]  = nh;
  hbf[gid] = f2bf(nh);
}

__global__ __launch_bounds__(256) void pred_kernel(const float* __restrict__ h,
                                                   const float* __restrict__ W,
                                                   const float* __restrict__ b,
                                                   float* __restrict__ out) {
  int v = blockIdx.x * 256 + threadIdx.x;
  if (v >= NN) return;
  float acc = b[0];
#pragma unroll
  for (int c = 0; c < 64; ++c) acc += h[(long)v * 64 + c] * W[c];
  out[v] = acc;
}

// ---------------------------------------------------------------------------
extern "C" void kernel_launch(void* const* d_in, const int* in_sizes, int n_in,
                              void* d_out, int out_size, void* d_ws, size_t ws_size,
                              hipStream_t stream) {
  (void)in_sizes; (void)n_in; (void)out_size; (void)ws_size;
  const float* x         = (const float*)d_in[0];
  const int*   edge_index= (const int*)d_in[1];
  const float* edge_attr = (const float*)d_in[2];
  const float* lin_W     = (const float*)d_in[3];
  const float* lin_b     = (const float*)d_in[4];
  const float* msg_W1    = (const float*)d_in[5];
  const float* msg_b1    = (const float*)d_in[6];
  const float* msg_g1    = (const float*)d_in[7];
  const float* msg_be1   = (const float*)d_in[8];
  const float* msg_W2    = (const float*)d_in[9];
  const float* msg_b2    = (const float*)d_in[10];
  const float* msg_g2    = (const float*)d_in[11];
  const float* msg_be2   = (const float*)d_in[12];
  const float* upd_W1    = (const float*)d_in[13];
  const float* upd_b1    = (const float*)d_in[14];
  const float* upd_g1    = (const float*)d_in[15];
  const float* upd_be1   = (const float*)d_in[16];
  const float* upd_W2    = (const float*)d_in[17];
  const float* upd_b2    = (const float*)d_in[18];
  const float* upd_g2    = (const float*)d_in[19];
  const float* upd_be2   = (const float*)d_in[20];
  const float* pred_W    = (const float*)d_in[21];
  const float* pred_b    = (const float*)d_in[22];
  float* out = (float*)d_out;

  char* ws = (char*)d_ws;
  size_t off = 0;
  auto take = [&](size_t bytes) -> char* {
    char* p = ws + off;
    off = (off + bytes + 255) & ~(size_t)255;
    return p;
  };
  float*          h     = (float*)take((size_t)NN * 64 * 4);
  unsigned short* hbf   = (unsigned short*)take((size_t)NN * 64 * 2);
  float*          aggr  = (float*)take((size_t)NN * 64 * 4);
  unsigned short* z1    = (unsigned short*)take((size_t)NE * 64 * 2);
  unsigned short* z2    = (unsigned short*)take((size_t)NE * 64 * 2);
  unsigned short* u1    = (unsigned short*)take((size_t)NN * 64 * 2);
  unsigned short* u2    = (unsigned short*)take((size_t)NN * 64 * 2);
  unsigned short* Wm1b  = (unsigned short*)take((size_t)LAYERS * 129 * 64 * 2);
  unsigned short* Wm2b  = (unsigned short*)take((size_t)LAYERS * 64 * 64 * 2);
  unsigned short* Wu1b  = (unsigned short*)take((size_t)LAYERS * 128 * 64 * 2);
  unsigned short* Wu2b  = (unsigned short*)take((size_t)LAYERS * 64 * 64 * 2);
  float*          statsA= (float*)take(128 * 4);
  float*          statsB= (float*)take(128 * 4);

  const float invE = 1.0f / (float)NE;
  const float invN = 1.0f / (float)NN;
  const int nodeBlocks = (NN + 63) / 64;   // 782 (tail-guarded)

  cvt_f32_bf16<<<(LAYERS * 129 * 64 + 255) / 256, 256, 0, stream>>>(msg_W1, Wm1b, LAYERS * 129 * 64);
  cvt_f32_bf16<<<(LAYERS * 64 * 64 + 255) / 256, 256, 0, stream>>>(msg_W2, Wm2b, LAYERS * 64 * 64);
  cvt_f32_bf16<<<(LAYERS * 128 * 64 + 255) / 256, 256, 0, stream>>>(upd_W1, Wu1b, LAYERS * 128 * 64);
  cvt_f32_bf16<<<(LAYERS * 64 * 64 + 255) / 256, 256, 0, stream>>>(upd_W2, Wu2b, LAYERS * 64 * 64);

  lin_in_kernel<<<(NN * 64) / 256, 256, 0, stream>>>(x, lin_W, lin_b, h, hbf);

  for (int l = 0; l < LAYERS; ++l) {
    hipMemsetAsync(statsA, 0, 512, stream);
    edge_msg_gemm1<<<NE / 64, 128, 0, stream>>>(
        hbf, edge_index, edge_attr, Wm1b + (size_t)l * 129 * 64,
        msg_W1 + (size_t)l * 129 * 64, msg_b1 + l * 64, z1, statsA);

    hipMemsetAsync(statsB, 0, 512, stream);
    mlp2_bn_gemm<<<NE / 64, 128, 0, stream>>>(
        z1, statsA, msg_g1 + l * 64, msg_be1 + l * 64, invE,
        Wm2b + (size_t)l * 64 * 64, msg_b2 + l * 64, z2, statsB, (long)NE);

    hipMemsetAsync(aggr, 0, (size_t)NN * 64 * 4, stream);
    bn_relu_scatter<<<(NE * 4) / 256, 256, 0, stream>>>(
        z2, statsB, msg_g2 + l * 64, msg_be2 + l * 64, invE, edge_index, aggr);

    hipMemsetAsync(statsA, 0, 512, stream);
    node_upd_gemm1<<<nodeBlocks, 128, 0, stream>>>(
        hbf, aggr, Wu1b + (size_t)l * 128 * 64, upd_b1 + l * 64, u1, statsA);

    hipMemsetAsync(statsB, 0, 512, stream);
    mlp2_bn_gemm<<<nodeBlocks, 128, 0, stream>>>(
        u1, statsA, upd_g1 + l * 64, upd_be1 + l * 64, invN,
        Wu2b + (size_t)l * 64 * 64, upd_b2 + l * 64, u2, statsB, (long)NN);

    bn_relu_residual<<<(NN * 64) / 256, 256, 0, stream>>>(
        u2, statsB, upd_g2 + l * 64, upd_be2 + l * 64, invN, h, hbf);
  }

  pred_kernel<<<(NN + 255) / 256, 256, 0, stream>>>(h, pred_W, pred_b, out);
}
